// LSTMTagger_3882650436249
// MI455X (gfx1250) — matrix operand order repaired and checked by reading, not compiled, not run on hardware
//
#include <hip/hip_runtime.h>
#include <hip/hip_bf16.h>
#include <math.h>

// Problem dims (from reference)
#define S_WORDS 4096
#define L_CHARS 16
#define CHAR_DIM 128
#define CHAR_H   256
#define CH4      1024     // 4*CHAR_H
#define N_DIM    512
#define N_HID    1024
#define WH4      4096     // 4*N_HID
#define FEAT     768      // N_DIM + CHAR_H
#define N_TAG    64

#define TM 4              // 16-row tiles per wave (64 rows)
#define TN 2              // 16-col tiles per wave (32 cols)

typedef __attribute__((ext_vector_type(16))) __bf16 bf16x16;
typedef __attribute__((ext_vector_type(8)))  float  fx8;

__device__ __forceinline__ float sigm_(float x){ return 1.0f/(1.0f+__expf(-x)); }
__device__ __forceinline__ float bf_lo2f(unsigned u){ union{unsigned i;float f;}v; v.i = u<<16;         return v.f; }
__device__ __forceinline__ float bf_hi2f(unsigned u){ union{unsigned i;float f;}v; v.i = u&0xffff0000u; return v.f; }

// ---------------- elementwise / setup kernels ----------------
__global__ void k_f32_to_bf16(const float* __restrict__ src, __hip_bfloat16* __restrict__ dst, int n){
  int i = blockIdx.x*blockDim.x + threadIdx.x;
  if (i < n) dst[i] = __float2bfloat16(src[i]);
}
__global__ void k_zero_f32(float* __restrict__ p, int n){
  int i = blockIdx.x*blockDim.x + threadIdx.x;
  if (i < n) p[i] = 0.0f;
}
__global__ void k_zero_bf16(__hip_bfloat16* __restrict__ p, int n){
  int i = blockIdx.x*blockDim.x + threadIdx.x;
  if (i < n) p[i] = __float2bfloat16(0.0f);
}
// dst[row*stride + off + col] = bf16(table[idx[row]*dim + col])
__global__ void k_gather_bf16(const int* __restrict__ idx, const float* __restrict__ table,
                              int dim, __hip_bfloat16* __restrict__ dst, int dst_stride, int dst_off,
                              long n){
  long i = (long)blockIdx.x*blockDim.x + threadIdx.x;
  if (i >= n) return;
  long row = i / dim, col = i - row*dim;
  dst[row*(long)dst_stride + dst_off + col] =
      __float2bfloat16(table[(long)idx[row]*dim + col]);
}

// ---------------- WMMA GEMM: C[M,N] = A[M,K] * W[N,K]^T ----------------
// A bf16 row-major (LDA compile-time); W bf16 row-major [N,K] (== B^T: the
// B-fragment load is a contiguous row of W). Each wave computes a 64x32 output
// block (4x2 tiles of 16x16); K unrolled by 64 (two k-blocks per iteration) so
// the second block's loads overlap the first block's 8 WMMAs. Compile-time
// LDA/LDW fold all tile offsets into immediate offsets: only two live address
// pointers in the loop -> no VGPR spills.
// Requires M%64==0, N%32==0, K%64==0 (true for all calls here).
template<int LDA, int LDW>
__device__ __forceinline__ void gemm_tiles_t(const __hip_bfloat16* __restrict__ A,
                                             const __hip_bfloat16* __restrict__ W,
                                             int K, int m0, int n0, int lane,
                                             fx8 acc[TM][TN]){
  const int rc   = lane & 15;   // A-row within tile / B-col within tile
  const int half = lane >> 4;
  const __bf16* ap = reinterpret_cast<const __bf16*>(A) + (size_t)(m0 + rc)*LDA + half*16;
  const __bf16* bp = reinterpret_cast<const __bf16*>(W) + (size_t)(n0 + rc)*LDW + half*16;
  for (int k0 = 0; k0 < K; k0 += 64, ap += 64, bp += 64){
    bf16x16 a0[TM], a1[TM], b0[TN], b1[TN];
#pragma unroll
    for (int mi = 0; mi < TM; ++mi){
#pragma unroll
      for (int e = 0; e < 8; ++e){
        a0[mi][e]   = ap[mi*16*LDA + e];
        a0[mi][e+8] = ap[mi*16*LDA + 8 + e];
        a1[mi][e]   = ap[mi*16*LDA + 32 + e];
        a1[mi][e+8] = ap[mi*16*LDA + 40 + e];
      }
    }
#pragma unroll
    for (int ni = 0; ni < TN; ++ni){
#pragma unroll
      for (int e = 0; e < 16; ++e){
        b0[ni][e] = bp[ni*16*LDW + e];
        b1[ni][e] = bp[ni*16*LDW + 32 + e];
      }
    }
#pragma unroll
    for (int mi = 0; mi < TM; ++mi)
#pragma unroll
      for (int ni = 0; ni < TN; ++ni)
        acc[mi][ni] = __builtin_amdgcn_wmma_f32_16x16x32_bf16(
            false, a0[mi], false, b0[ni], (short)0, acc[mi][ni], false, false);
#pragma unroll
    for (int mi = 0; mi < TM; ++mi)
#pragma unroll
      for (int ni = 0; ni < TN; ++ni)
        acc[mi][ni] = __builtin_amdgcn_wmma_f32_16x16x32_bf16(
            false, a1[mi], false, b1[ni], (short)0, acc[mi][ni], false, false);
  }
}

template<int LDA, int LDW>
__global__ __launch_bounds__(128, 1)
void k_gemm_f32out(const __hip_bfloat16* __restrict__ A,
                   const __hip_bfloat16* __restrict__ W,
                   float* __restrict__ C, int ldc,
                   int M, int N, int K, int accumulate){
  int wave = blockIdx.x*(blockDim.x >> 5) + (threadIdx.x >> 5);
  int lane = threadIdx.x & 31;
  int stn = N >> 5;                       // 32-wide super-tiles in N
  int total = (M >> 6)*stn;               // 64-tall super-tiles in M
  if (wave >= total) return;              // wave-uniform: EXEC stays all-1s
  int m0 = (wave / stn) << 6;
  int n0 = (wave % stn) << 5;
  int col = lane & 15, half = lane >> 4;  // C tile: row = half*8 + r, col = lane&15
  fx8 acc[TM][TN];
  if (accumulate){
#pragma unroll
    for (int mi = 0; mi < TM; ++mi)
#pragma unroll
      for (int ni = 0; ni < TN; ++ni)
#pragma unroll
        for (int r = 0; r < 8; ++r)
          acc[mi][ni][r] = C[(size_t)(m0 + mi*16 + half*8 + r)*ldc + n0 + ni*16 + col];
  } else {
#pragma unroll
    for (int mi = 0; mi < TM; ++mi)
#pragma unroll
      for (int ni = 0; ni < TN; ++ni)
#pragma unroll
        for (int r = 0; r < 8; ++r) acc[mi][ni][r] = 0.0f;
  }
  gemm_tiles_t<LDA, LDW>(A, W, K, m0, n0, lane, acc);
#pragma unroll
  for (int mi = 0; mi < TM; ++mi)
#pragma unroll
    for (int ni = 0; ni < TN; ++ni)
#pragma unroll
      for (int r = 0; r < 8; ++r)
        C[(size_t)(m0 + mi*16 + half*8 + r)*ldc + n0 + ni*16 + col] = acc[mi][ni][r];
}

template<int LDA, int LDW>
__global__ __launch_bounds__(128, 1)
void k_gemm_bf16out(const __hip_bfloat16* __restrict__ A,
                    const __hip_bfloat16* __restrict__ W,
                    __hip_bfloat16* __restrict__ C, int ldc,
                    int M, int N, int K){
  int wave = blockIdx.x*(blockDim.x >> 5) + (threadIdx.x >> 5);
  int lane = threadIdx.x & 31;
  int stn = N >> 5;
  int total = (M >> 6)*stn;
  if (wave >= total) return;
  int m0 = (wave / stn) << 6;
  int n0 = (wave % stn) << 5;
  int col = lane & 15, half = lane >> 4;
  fx8 acc[TM][TN];
#pragma unroll
  for (int mi = 0; mi < TM; ++mi)
#pragma unroll
    for (int ni = 0; ni < TN; ++ni)
#pragma unroll
      for (int r = 0; r < 8; ++r) acc[mi][ni][r] = 0.0f;
  gemm_tiles_t<LDA, LDW>(A, W, K, m0, n0, lane, acc);
#pragma unroll
  for (int mi = 0; mi < TM; ++mi)
#pragma unroll
    for (int ni = 0; ni < TN; ++ni)
#pragma unroll
      for (int r = 0; r < 8; ++r)
        C[(size_t)(m0 + mi*16 + half*8 + r)*ldc + n0 + ni*16 + col] =
            __float2bfloat16(acc[mi][ni][r]);
}

// ---------------- char-LSTM pointwise step ----------------
// G[s,0:1024] = x_t@Wih^T + h@Whh^T (bias added here). Gate order i,f,g,o.
__global__ void k_char_pointwise(const float* __restrict__ G, const float* __restrict__ cb,
                                 float* __restrict__ c, __hip_bfloat16* __restrict__ h_next,
                                 __hip_bfloat16* __restrict__ feats, int is_last){
  int idx = blockIdx.x*blockDim.x + threadIdx.x;     // S_WORDS*CHAR_H
  if (idx >= S_WORDS*CHAR_H) return;
  int s = idx >> 8, j = idx & 255;
  const float* g = G + (size_t)s*CH4;
  float gi = g[j]       + cb[j];
  float gf = g[256 + j] + cb[256 + j];
  float gg = g[512 + j] + cb[512 + j];
  float go = g[768 + j] + cb[768 + j];
  float cn = sigm_(gf)*c[idx] + sigm_(gi)*tanhf(gg);
  float h  = sigm_(go)*tanhf(cn);
  c[idx] = cn;
  h_next[(size_t)s*CHAR_H + j] = __float2bfloat16(h);
  if (is_last) feats[(size_t)s*FEAT + N_DIM + j] = __float2bfloat16(h);
}

// ---------------- word-LSTM sequential step ----------------
// One launch per timestep. 4 blocks x 256 threads; thread j computes all four
// gate dot-products over h (cached in LDS). Whh in bf16: 8MB/step resident in
// the 192MB L2 across all 4096 steps; fp32 accumulation.
__global__ void k_word_step(const __hip_bfloat16* __restrict__ Gih_row, // [WH4]
                            const __hip_bfloat16* __restrict__ Whh_bf,  // [WH4, N_HID] bf16
                            const float* __restrict__ wb,               // [WH4]
                            const float* __restrict__ h_in,             // [N_HID]
                            float* __restrict__ h_out,                  // [N_HID]
                            float* __restrict__ c,                      // [N_HID]
                            __hip_bfloat16* __restrict__ hs_row){       // [N_HID]
  __shared__ float sh[N_HID];
  int tid = threadIdx.x;
  for (int i = tid; i < N_HID; i += blockDim.x) sh[i] = h_in[i];
  __syncthreads();
  int j = blockIdx.x*blockDim.x + tid;               // 0..1023
  // speculative prefetch of next timestep's gate row (result-neutral)
  __builtin_prefetch((const char*)(Gih_row + WH4) + j*8, 0, 0);
  float acc[4];
#pragma unroll
  for (int gq = 0; gq < 4; ++gq){
    int gi = gq*N_HID + j;
    const uint4* wr = reinterpret_cast<const uint4*>(Whh_bf + (size_t)gi*N_HID); // 8 bf16 / 16B
    float s0 = 0.f, s1 = 0.f;
    for (int k = 0; k < N_HID/8; ++k){
      uint4 w = wr[k];
      const float* hp = sh + k*8;
      s0 += bf_lo2f(w.x)*hp[0] + bf_hi2f(w.x)*hp[1]
          + bf_lo2f(w.y)*hp[2] + bf_hi2f(w.y)*hp[3];
      s1 += bf_lo2f(w.z)*hp[4] + bf_hi2f(w.z)*hp[5]
          + bf_lo2f(w.w)*hp[6] + bf_hi2f(w.w)*hp[7];
    }
    acc[gq] = __bfloat162float(Gih_row[gi]) + wb[gi] + s0 + s1;
  }
  float cn = sigm_(acc[1])*c[j] + sigm_(acc[0])*tanhf(acc[2]);
  float h  = sigm_(acc[3])*tanhf(cn);
  c[j] = cn;
  h_out[j] = h;
  hs_row[j] = __float2bfloat16(h);
}

// ---------------- log-softmax over 64 tags ----------------
__global__ void k_logsoftmax(const float* __restrict__ logits, const float* __restrict__ linb,
                             float* __restrict__ out){
  __shared__ float red[N_TAG];
  int row = blockIdx.x, tid = threadIdx.x;
  float v = logits[(size_t)row*N_TAG + tid] + linb[tid];
  red[tid] = v; __syncthreads();
  for (int s = 32; s > 0; s >>= 1){ if (tid < s) red[tid] = fmaxf(red[tid], red[tid+s]); __syncthreads(); }
  float m = red[0]; __syncthreads();
  red[tid] = __expf(v - m); __syncthreads();
  for (int s = 32; s > 0; s >>= 1){ if (tid < s) red[tid] += red[tid+s]; __syncthreads(); }
  float lse = logf(red[0]);
  out[(size_t)row*N_TAG + tid] = v - m - lse;
}

// ---------------- host-side orchestration ----------------
static inline int gemm_blocks(int M, int N){
  int waves = (M >> 6)*(N >> 5);           // 64x32 output per wave
  return (waves + 3) >> 2;                 // 4 waves (128 threads) per block
}

extern "C" void kernel_launch(void* const* d_in, const int* in_sizes, int n_in,
                              void* d_out, int out_size, void* d_ws, size_t ws_size,
                              hipStream_t stream) {
  const int*   x        = (const int*)  d_in[0];
  const int*   chars    = (const int*)  d_in[1];
  const float* word_emb = (const float*)d_in[2];
  const float* char_emb = (const float*)d_in[3];
  const float* cW_ih    = (const float*)d_in[4];
  const float* cW_hh    = (const float*)d_in[5];
  const float* cb       = (const float*)d_in[6];
  const float* wW_ih    = (const float*)d_in[7];
  const float* wW_hh    = (const float*)d_in[8];
  const float* wb       = (const float*)d_in[9];
  const float* linW     = (const float*)d_in[10];
  const float* linb     = (const float*)d_in[11];
  float* out = (float*)d_out;

  // carve workspace
  char* ws = (char*)d_ws;
  size_t off = 0;
  auto carve = [&](size_t bytes)->void*{ void* p = ws + off; off += (bytes + 255) & ~(size_t)255; return p; };

  __hip_bfloat16* char_x_bf = (__hip_bfloat16*)carve((size_t)S_WORDS*L_CHARS*CHAR_DIM*2); // 16MB
  __hip_bfloat16* cWih_bf   = (__hip_bfloat16*)carve((size_t)CH4*CHAR_DIM*2);
  __hip_bfloat16* cWhh_bf   = (__hip_bfloat16*)carve((size_t)CH4*CHAR_H*2);
  __hip_bfloat16* wWih_bf   = (__hip_bfloat16*)carve((size_t)WH4*FEAT*2);                 // 6MB
  __hip_bfloat16* wWhh_bf   = (__hip_bfloat16*)carve((size_t)WH4*N_HID*2);                // 8MB
  __hip_bfloat16* linW_bf   = (__hip_bfloat16*)carve((size_t)N_TAG*N_HID*2);
  float*          Gc        = (float*)carve((size_t)S_WORDS*CH4*4);                       // 16MB
  float*          c_char    = (float*)carve((size_t)S_WORDS*CHAR_H*4);                    // 4MB
  __hip_bfloat16* hA        = (__hip_bfloat16*)carve((size_t)S_WORDS*CHAR_H*2);
  __hip_bfloat16* hB        = (__hip_bfloat16*)carve((size_t)S_WORDS*CHAR_H*2);
  __hip_bfloat16* feats_bf  = (__hip_bfloat16*)carve((size_t)S_WORDS*FEAT*2);             // 6MB
  __hip_bfloat16* Gw_bf     = (__hip_bfloat16*)carve((size_t)S_WORDS*WH4*2);              // 32MB
  float*          hw0       = (float*)carve((size_t)N_HID*4);
  float*          hw1       = (float*)carve((size_t)N_HID*4);
  float*          cw        = (float*)carve((size_t)N_HID*4);
  __hip_bfloat16* hs_bf     = (__hip_bfloat16*)carve((size_t)S_WORDS*N_HID*2);            // 8MB
  float*          logits    = (float*)carve((size_t)S_WORDS*N_TAG*4);                     // 1MB
  (void)ws_size; (void)in_sizes; (void)n_in; (void)out_size;

  const int TB = 256;
  auto nb = [&](long n){ return (int)((n + TB - 1)/TB); };

  // weight conversions to bf16
  k_f32_to_bf16<<<nb(CH4*CHAR_DIM), TB, 0, stream>>>(cW_ih, cWih_bf, CH4*CHAR_DIM);
  k_f32_to_bf16<<<nb(CH4*CHAR_H),   TB, 0, stream>>>(cW_hh, cWhh_bf, CH4*CHAR_H);
  k_f32_to_bf16<<<nb(WH4*FEAT),     TB, 0, stream>>>(wW_ih, wWih_bf, WH4*FEAT);
  k_f32_to_bf16<<<nb(WH4*N_HID),    TB, 0, stream>>>(wW_hh, wWhh_bf, WH4*N_HID);
  k_f32_to_bf16<<<nb(N_TAG*N_HID),  TB, 0, stream>>>(linW,  linW_bf, N_TAG*N_HID);

  // embedding gathers (straight to bf16)
  k_gather_bf16<<<nb((long)S_WORDS*L_CHARS*CHAR_DIM), TB, 0, stream>>>(
      chars, char_emb, CHAR_DIM, char_x_bf, CHAR_DIM, 0, (long)S_WORDS*L_CHARS*CHAR_DIM);
  k_gather_bf16<<<nb((long)S_WORDS*N_DIM), TB, 0, stream>>>(
      x, word_emb, N_DIM, feats_bf, FEAT, 0, (long)S_WORDS*N_DIM);

  // init states
  k_zero_f32 <<<nb(S_WORDS*CHAR_H), TB, 0, stream>>>(c_char, S_WORDS*CHAR_H);
  k_zero_bf16<<<nb(S_WORDS*CHAR_H), TB, 0, stream>>>(hA, S_WORDS*CHAR_H);
  k_zero_f32 <<<nb(N_HID), TB, 0, stream>>>(hw0, N_HID);
  k_zero_f32 <<<nb(N_HID), TB, 0, stream>>>(cw,  N_HID);

  // ---- char LSTM: 16 steps, each = (ih GEMM) + (hh GEMM accum) + pointwise
  __hip_bfloat16* h_prev = hA;
  __hip_bfloat16* h_next = hB;
  for (int t = 0; t < L_CHARS; ++t){
    // G = char_x[:,t,:] @ cW_ih^T  (A slice: base + t*CHAR_DIM, lda = L*CHAR_DIM = 2048)
    k_gemm_f32out<L_CHARS*CHAR_DIM, CHAR_DIM><<<gemm_blocks(S_WORDS, CH4), 128, 0, stream>>>(
        char_x_bf + (size_t)t*CHAR_DIM, cWih_bf, Gc, CH4, S_WORDS, CH4, CHAR_DIM, 0);
    if (t > 0){
      // G += h_prev @ cW_hh^T
      k_gemm_f32out<CHAR_H, CHAR_H><<<gemm_blocks(S_WORDS, CH4), 128, 0, stream>>>(
          h_prev, cWhh_bf, Gc, CH4, S_WORDS, CH4, CHAR_H, 1);
    }
    k_char_pointwise<<<nb(S_WORDS*CHAR_H), TB, 0, stream>>>(
        Gc, cb, c_char, h_next, feats_bf, (t == L_CHARS-1) ? 1 : 0);
    __hip_bfloat16* tmp = h_prev; h_prev = h_next; h_next = tmp;
  }

  // ---- word LSTM input projection: Gw = feats @ wW_ih^T  [4096 x 4096]
  k_gemm_bf16out<FEAT, FEAT><<<gemm_blocks(S_WORDS, WH4), 128, 0, stream>>>(
      feats_bf, wWih_bf, Gw_bf, WH4, S_WORDS, WH4, FEAT);

  // ---- word LSTM recurrence: 4096 sequential mat-vec steps (L2-resident Whh)
  float* hin = hw0; float* hout = hw1;
  for (int t = 0; t < S_WORDS; ++t){
    k_word_step<<<4, 256, 0, stream>>>(
        Gw_bf + (size_t)t*WH4, wWhh_bf, wb, hin, hout, cw, hs_bf + (size_t)t*N_HID);
    float* tmp = hin; hin = hout; hout = tmp;
  }

  // ---- final linear + log-softmax
  k_gemm_f32out<N_HID, N_HID><<<gemm_blocks(S_WORDS, N_TAG), 128, 0, stream>>>(
      hs_bf, linW_bf, logits, N_TAG, S_WORDS, N_TAG, N_HID, 0);
  k_logsoftmax<<<S_WORDS, N_TAG, 0, stream>>>(logits, linb, out);
}
